// TOCCA1_41025527611545
// MI455X (gfx1250) — compile-verified, matrix operand order checked by reading, not run on hardware
//
#include <hip/hip_runtime.h>

// ---------------------------------------------------------------------------
// GCN fusion for MI455X (gfx1250, wave32, WMMA).
// Dominant cost: 4 edge aggregations (~3.3 GB of L2 traffic, L2-resident at
// 192MB). GEMMs (4 GFLOP total) are fused with all scaling/activation/BN work
// using v_wmma_f32_16x16x32_bf16 so every N x 64 intermediate is written once.
// GEMM is templated on its fusion mode so all feature flags & tile loops are
// compile-time (previous round showed scratch spills from runtime branches).
// ---------------------------------------------------------------------------

typedef __attribute__((ext_vector_type(16))) __bf16 v16bf;
typedef __attribute__((ext_vector_type(8)))  __bf16 v8bf;
typedef __attribute__((ext_vector_type(8)))  float  v8f;

#define THREADS 256

__device__ __forceinline__ long lmin(long a, long b) { return a < b ? a : b; }

__device__ __forceinline__ v8f wmma_bf16(v16bf a, v16bf b, v8f c) {
  // 8 args: (neg_a, A, neg_b, B, c_mod, C, reuse_a, reuse_b)
  return __builtin_amdgcn_wmma_f32_16x16x32_bf16(false, a, false, b,
                                                 (short)0, c, false, false);
}

__device__ __forceinline__ v16bf cat16(v8bf lo, v8bf hi) {
  return __builtin_shufflevector(lo, hi, 0, 1, 2, 3, 4, 5, 6, 7,
                                 8, 9, 10, 11, 12, 13, 14, 15);
}

// -------------------------------- utility ----------------------------------

__global__ void zero_k(float* __restrict__ p, long n) {
  long i = (long)blockIdx.x * blockDim.x + threadIdx.x;
  long stride = (long)gridDim.x * blockDim.x;
  for (; i < n; i += stride) p[i] = 0.f;
}

__global__ void degree_k(const int* __restrict__ src, const int* __restrict__ dst,
                         float* __restrict__ dout, float* __restrict__ din, int E) {
  int e = blockIdx.x * THREADS + threadIdx.x;
  if (e >= E) return;
  atomicAdd(&dout[src[e]], 1.f);
  atomicAdd(&din[dst[e]], 1.f);
}

__global__ void invsqrt_k(float* __restrict__ p, int n) {
  int i = blockIdx.x * THREADS + threadIdx.x;
  if (i < n) p[i] = rsqrtf(fmaxf(p[i], 1.f));
}

__global__ void rowscale_k(const float* __restrict__ x, const float* __restrict__ s,
                           float* __restrict__ y, long total) {
  long i = (long)blockIdx.x * THREADS + threadIdx.x;
  if (i < total) y[i] = x[i] * s[i >> 6];
}

// ----------------------------- edge scatter --------------------------------
// 16 threads per edge, float4 per thread: one fully coalesced 256B row read,
// 64 contiguous f32 atomics into the destination row (L2 atomic units).

__global__ void scatter_k(const int* __restrict__ src, const int* __restrict__ dst,
                          const float* __restrict__ X, float* __restrict__ Y, long E) {
  long t = (long)blockIdx.x * THREADS + threadIdx.x;
  long e = t >> 4;
  if (e >= E) return;
  int part = (int)(t & 15);
  long s = src[e], d = dst[e];
  const float4 v = *(const float4*)(X + s * 64 + part * 4);
  float* p = Y + d * 64 + part * 4;
  atomicAdd(p + 0, v.x); atomicAdd(p + 1, v.y);
  atomicAdd(p + 2, v.z); atomicAdd(p + 3, v.w);
}

// ------------------------------ fused GEMM ---------------------------------
// Out = postOps( prologue(X) @ W + bias ),  X: N x 64, W: 64 x NCOL.
//   PRE : per-row preScale on input      (GraphConv in-degree norm)
//   BN  : per-k BatchNorm+ReLU on input  (MLP head)
//   RELU: ReLU on output                 (GraphConv layer 1)
//   POST: per-row postScale on output    (pre-applies next layer's src norm)
// Block = 256 threads = 8 waves, wave owns a 16-row strip, K=64 = 2 k-steps.
// W staged transposed in LDS (padded rows) so B-fragments are ds_load_b128s.

template <int NCOL, bool PRE, bool POST, bool RELU, bool BN>
__global__ void __launch_bounds__(256)
gemm64_t(const float* __restrict__ X, const float* __restrict__ W,
         const float* __restrict__ bias, float* __restrict__ Out,
         const float* __restrict__ preScale, const float* __restrict__ postScale,
         const float* __restrict__ kmu, const float* __restrict__ kinv,
         const float* __restrict__ kgamma, const float* __restrict__ kbeta,
         int N)
{
  constexpr int TILES = (NCOL + 15) / 16;
  constexpr int CT16  = TILES * 16;
  constexpr int LDW   = 80;  // bf16 row stride: 160B rows -> 16B aligned, banks spread

  __shared__ __bf16 WlT[CT16 * LDW];  // WlT[n*LDW + k] = W[k][n] (bf16)
  __shared__ float  biasL[CT16];

  const int tid = threadIdx.x;
  for (int idx = tid; idx < CT16 * 64; idx += 256) {
    int n = idx >> 6, k = idx & 63;
    float w = (n < NCOL) ? W[k * NCOL + n] : 0.f;
    WlT[n * LDW + k] = (__bf16)w;
  }
  if (tid < CT16) biasL[tid] = (tid < NCOL) ? bias[tid] : 0.f;
  __syncthreads();

  const int  wave  = tid >> 5, lane = tid & 31;
  const long rowBase = (long)blockIdx.x * 128 + wave * 16;
  const int  mlane = lane & 15;
  const long row  = lmin(rowBase + mlane, (long)N - 1);
  const int  kg   = (lane < 16) ? 0 : 8;  // fragment lane-half K offset (ISA 7.12.2)
  const float* xr = X + row * 64;

  float ps = 1.f;
  if (PRE) ps = preScale[row];

  // A fragments (16x32 bf16): element i -> K = s*32 + kg + i (i<8), +8+i (i>=8)
  v16bf a[2];
  #pragma unroll
  for (int s = 0; s < 2; ++s) {
    const float4 x0 = *(const float4*)(xr + s * 32 + kg);
    const float4 x1 = *(const float4*)(xr + s * 32 + kg + 4);
    const float4 x2 = *(const float4*)(xr + s * 32 + kg + 16);
    const float4 x3 = *(const float4*)(xr + s * 32 + kg + 20);
    float xv[16] = {x0.x, x0.y, x0.z, x0.w, x1.x, x1.y, x1.z, x1.w,
                    x2.x, x2.y, x2.z, x2.w, x3.x, x3.y, x3.z, x3.w};
    #pragma unroll
    for (int i = 0; i < 16; ++i) {
      int k = s * 32 + kg + ((i < 8) ? i : (8 + i));
      float x = xv[i];
      if (BN) x = fmaxf((x - kmu[k]) * kinv[k] * kgamma[k] + kbeta[k], 0.f);
      if (PRE) x *= ps;
      a[s][i] = (__bf16)x;
    }
  }

  // hoist per-row output scale (8 rows per lane)
  float postv[8];
  if (POST) {
    #pragma unroll
    for (int v = 0; v < 8; ++v)
      postv[v] = postScale[lmin(rowBase + ((lane < 16) ? v : v + 8), (long)N - 1)];
  }

  const bool fullTile = (rowBase + 16) <= (long)N;  // wave-uniform

  #pragma unroll
  for (int c = 0; c < TILES; ++c) {
    const int n = mlane + c * 16;
    const __bf16* wr = &WlT[n * LDW];
    v16bf b0 = cat16(*(const v8bf*)(wr + kg),      *(const v8bf*)(wr + kg + 16));
    v16bf b1 = cat16(*(const v8bf*)(wr + kg + 32), *(const v8bf*)(wr + kg + 48));

    v8f acc = {};
    acc = wmma_bf16(a[0], b0, acc);
    acc = wmma_bf16(a[1], b1, acc);

    const float bn = biasL[n];
    float vals[8];
    #pragma unroll
    for (int v = 0; v < 8; ++v) {
      float val = acc[v] + bn;
      if (RELU) val = fmaxf(val, 0.f);
      if (POST) val *= postv[v];
      vals[v] = val;
    }

    float* op = Out + (rowBase + ((lane < 16) ? 0 : 8)) * NCOL + n;
    if (fullTile) {
      if (NCOL % 16 == 0 || n < NCOL) {
        #pragma unroll
        for (int v = 0; v < 8; ++v) op[(long)v * NCOL] = vals[v];
      }
    } else {
      #pragma unroll
      for (int v = 0; v < 8; ++v) {
        long r = rowBase + ((lane < 16) ? v : v + 8);
        if (r < N && (NCOL % 16 == 0 || n < NCOL)) Out[r * NCOL + n] = vals[v];
      }
    }
  }
}

// --------------------------- column statistics -----------------------------

__global__ void colstats_k(const float* __restrict__ H, float* __restrict__ sum,
                           float* __restrict__ sq, int N) {
  int c = threadIdx.x & 63;   // column (coalesced across lanes)
  int rg = threadIdx.x >> 6;  // 0..3 row groups per block
  float s = 0.f, q = 0.f;
  for (long r = (long)blockIdx.x * 4 + rg; r < N; r += (long)gridDim.x * 4) {
    float v = H[r * 64 + c];
    s += v; q += v * v;
  }
  __shared__ float Ss[256], Sq[256];
  Ss[threadIdx.x] = s; Sq[threadIdx.x] = q;
  __syncthreads();
  if (rg == 0) {
    for (int i = 1; i < 4; ++i) { s += Ss[c + 64 * i]; q += Sq[c + 64 * i]; }
    atomicAdd(&sum[c], s); atomicAdd(&sq[c], q);
  }
}

__global__ void finalize_k(const float* __restrict__ sum, const float* __restrict__ sq,
                           float* __restrict__ mu, float* __restrict__ inv,
                           float n, int bn) {
  int c = threadIdx.x;
  if (c >= 64) return;
  float m = sum[c] / n;
  float var = bn ? (sq[c] / n - m * m + 1e-5f)          // biased + eps (BN)
                 : ((sq[c] - n * m * m) / (n - 1.f));   // unbiased (torch .std)
  mu[c] = m; inv[c] = rsqrtf(var);
}

// standardize z1/z2 in place in d_out and emit Z = 0.5*(z1+z2)
__global__ void stdcomb_k(float* __restrict__ Z1, float* __restrict__ Z2,
                          float* __restrict__ Z,
                          const float* __restrict__ mu1, const float* __restrict__ inv1,
                          const float* __restrict__ mu2, const float* __restrict__ inv2,
                          long total) {
  long i = (long)blockIdx.x * THREADS + threadIdx.x;
  if (i >= total) return;
  int c = (int)(i & 63);
  float a = (Z1[i] - mu1[c]) * inv1[c];
  float b = (Z2[i] - mu2[c]) * inv2[c];
  Z1[i] = a; Z2[i] = b; Z[i] = 0.5f * (a + b);
}

// ------------------------------- launcher ----------------------------------

extern "C" void kernel_launch(void* const* d_in, const int* in_sizes, int n_in,
                              void* d_out, int out_size, void* d_ws, size_t ws_size,
                              hipStream_t stream) {
  const int N = in_sizes[0] / 64;
  const int E = in_sizes[1];

  const float* feat[2] = {(const float*)d_in[0], (const float*)d_in[3]};
  const int*   src[2]  = {(const int*)d_in[1],   (const int*)d_in[4]};
  const int*   dst[2]  = {(const int*)d_in[2],   (const int*)d_in[5]};
  const float* W1[2]   = {(const float*)d_in[6], (const float*)d_in[10]};
  const float* b1[2]   = {(const float*)d_in[7], (const float*)d_in[11]};
  const float* W2[2]   = {(const float*)d_in[8], (const float*)d_in[12]};
  const float* b2[2]   = {(const float*)d_in[9], (const float*)d_in[13]};
  const float* Wm1 = (const float*)d_in[14];
  const float* bm1 = (const float*)d_in[15];
  const float* gamma = (const float*)d_in[16];
  const float* beta  = (const float*)d_in[17];
  const float* Wm2 = (const float*)d_in[18];
  const float* bm2 = (const float*)d_in[19];

  float* out = (float*)d_out;
  float* z[2] = {out, out + (long)N * 64};
  float* pred = out + (long)N * 128;

  float* ws = (float*)d_ws;
  float* ns = ws;                // N (out-degree norm), nd contiguous after it
  float* nd = ws + N;            // N (in-degree norm)
  float* stats = ws + 2L * N;    // 1024 floats of stats scratch
  float* ssum = stats,       *ssq  = stats + 64;
  float* mu1  = stats + 128, *inv1 = stats + 192;
  float* mu2  = stats + 256, *inv2 = stats + 320;
  float* muY  = stats + 384, *invY = stats + 448;
  float* B0 = stats + 1024;      // N x 64 ping
  float* B1 = B0 + (long)N * 64; // N x 64 pong

  const long NF = (long)N * 64;
  const int gGemm = (N + 127) / 128;
  const int gElem = (int)((NF + THREADS - 1) / THREADS);
  const int gEdge = (E + THREADS - 1) / THREADS;
  const int gScat = (int)(((long)E * 16 + THREADS - 1) / THREADS);
  float* muG[2]  = {mu1, mu2};
  float* invG[2] = {inv1, inv2};

  for (int g = 0; g < 2; ++g) {
    // degree norms: clamp(deg,1)^-1/2
    zero_k<<<8, THREADS, 0, stream>>>(ns, 2L * N);
    degree_k<<<gEdge, THREADS, 0, stream>>>(src[g], dst[g], ns, nd, E);
    invsqrt_k<<<(2 * N + THREADS - 1) / THREADS, THREADS, 0, stream>>>(ns, 2 * N);

    // layer 1: scale -> scatter -> fused (agg*nd)@W1+b1 -> relu -> *ns
    rowscale_k<<<gElem, THREADS, 0, stream>>>(feat[g], ns, B0, NF);
    zero_k<<<1024, THREADS, 0, stream>>>(B1, NF);
    scatter_k<<<gScat, THREADS, 0, stream>>>(src[g], dst[g], B0, B1, E);
    gemm64_t<64, true, true, true, false><<<gGemm, THREADS, 0, stream>>>(
        B1, W1[g], b1[g], B0, nd, ns, nullptr, nullptr, nullptr, nullptr, N);

    // layer 2: scatter -> fused (agg*nd)@W2+b2 straight into d_out z-region
    zero_k<<<1024, THREADS, 0, stream>>>(B1, NF);
    scatter_k<<<gScat, THREADS, 0, stream>>>(src[g], dst[g], B0, B1, E);
    gemm64_t<64, true, false, false, false><<<gGemm, THREADS, 0, stream>>>(
        B1, W2[g], b2[g], z[g], nd, nullptr, nullptr, nullptr, nullptr, nullptr, N);

    // per-column mean / unbiased std
    zero_k<<<1, 128, 0, stream>>>(ssum, 128);
    colstats_k<<<512, THREADS, 0, stream>>>(z[g], ssum, ssq, N);
    finalize_k<<<1, 64, 0, stream>>>(ssum, ssq, muG[g], invG[g], (float)N, 0);
  }

  // z1/z2 in place + Z = (z1+z2)/2
  stdcomb_k<<<gElem, THREADS, 0, stream>>>(z[0], z[1], B0, mu1, inv1, mu2, inv2, NF);

  // MLP head: Y = Z@Wm1+bm1 ; BN stats ; pred = relu(BN(Y))@Wm2+bm2
  gemm64_t<64, false, false, false, false><<<gGemm, THREADS, 0, stream>>>(
      B0, Wm1, bm1, B1, nullptr, nullptr, nullptr, nullptr, nullptr, nullptr, N);
  zero_k<<<1, 128, 0, stream>>>(ssum, 128);
  colstats_k<<<512, THREADS, 0, stream>>>(B1, ssum, ssq, N);
  finalize_k<<<1, 64, 0, stream>>>(ssum, ssq, muY, invY, (float)N, 1);
  gemm64_t<40, false, false, false, true><<<gGemm, THREADS, 0, stream>>>(
      B1, Wm2, bm2, pred, nullptr, nullptr, muY, invY, gamma, beta, N);
}
